// ColorImplicitNetworkSingle_21766894256741
// MI455X (gfx1250) — matrix operand order repaired and checked by reading, not compiled, not run on hardware
//
#include <hip/hip_runtime.h>
#include <hip/hip_bf16.h>

// CDNA5 / gfx1250, wave32. Fused NGP hash-encode + 4-layer MLP.
// 512 threads (16 waves) per block; weights pre-converted to f16 images in
// d_ws by a prep kernel, then staged into LDS via the Tensor Data Mover
// (tensor_load_to_lds + s_wait_tensorcnt) instead of per-thread copies.

typedef __attribute__((ext_vector_type(16))) _Float16 v16h;
typedef __attribute__((ext_vector_type(8)))  _Float16 v8h;
typedef __attribute__((ext_vector_type(8)))  float    v8f;
typedef __attribute__((ext_vector_type(4)))  unsigned v4u;
typedef __attribute__((ext_vector_type(8)))  int      v8i;
typedef __attribute__((ext_vector_type(4)))  int      v4i;

#define SW 264   // LDS/image row stride in halves (132 dwords, %64==4 -> bank spread)

// ---- f16 weight images in d_ws (byte offsets, 8B aligned for TDM) ----
#define IMG1_OFF   0
#define IMG1_BYTES (32 * SW * 2)              // 16896
#define IMG2_OFF   17408
#define IMGW_BYTES (256 * SW * 2)             // 135168
#define IMG3_OFF   (IMG2_OFF + IMGW_BYTES)    // 152576
#define IMG4_OFF   (IMG3_OFF + IMGW_BYTES)    // 287744  (total ws: 422912 B)

// ---- hash-grid static config (L=16, D=2, base=16, desired=2048, maxp=2^19) ----
__device__ __constant__ float c_scale[16] = {
  15.0f, 21.110640f, 29.554874f, 41.224253f, 57.350212f, 79.634949f,
  110.430473f, 152.987198f, 211.796890f, 293.066772f, 405.374664f,
  560.574402f, 775.046875f, 1071.429199f, 1481.003662f, 2047.0f};
__device__ __constant__ unsigned c_offs[16] = {
  0u, 4920u, 18744u, 51512u, 136696u, 352696u, 876984u, 1401272u,
  1925560u, 2449848u, 2974136u, 3498424u, 4022712u, 4547000u,
  5071288u, 5595576u};
__device__ __constant__ unsigned c_dsize[5] = {4920u, 13824u, 32768u, 85184u, 216000u};
__device__ __constant__ unsigned c_res1[5]  = {17u, 24u, 32u, 44u, 60u};

// ================= prep kernel: f32 weights -> padded f16 images ===============
extern "C" __global__ void __launch_bounds__(256)
prep_weights(const float* __restrict__ w1, const float* __restrict__ w2,
             const float* __restrict__ w3, const float* __restrict__ w4,
             _Float16* __restrict__ ws)
{
  const int i = blockIdx.x * 256 + threadIdx.x;
  _Float16* img1 = ws + IMG1_OFF / 2;
  _Float16* img2 = ws + IMG2_OFF / 2;
  _Float16* img3 = ws + IMG3_OFF / 2;
  _Float16* img4 = ws + IMG4_OFF / 2;
  if (i < 8192) {                                        // w1: 32x256
    img1[(i >> 8) * SW + (i & 255)] = (_Float16)w1[i];
  } else if (i < 73728) {                                // w2: 256x256
    const int j = i - 8192;
    img2[(j >> 8) * SW + (j & 255)] = (_Float16)w2[j];
  } else if (i < 139264) {                               // w3: 256x256
    const int j = i - 73728;
    img3[(j >> 8) * SW + (j & 255)] = (_Float16)w3[j];
  } else if (i < 143360) {                               // w4: 256x3 zero-padded to 16
    const int j = i - 139264;
    const int r = j >> 4, c = j & 15;
    img4[r * SW + c] = (c < 3) ? (_Float16)w4[r * 3 + c] : (_Float16)0.0f;
  }
}

// ================= TDM staging: one DMA per layer ==============================
__device__ __forceinline__ void tdm_issue(const _Float16* gsrc, _Float16* ldst,
                                          unsigned n8 /* 8-byte units */) {
#if __has_builtin(__builtin_amdgcn_tensor_load_to_lds)
  const unsigned long long ga = (unsigned long long)(const void*)gsrc;
  const unsigned lo = (unsigned)(unsigned long long)(void*)ldst;  // LDS byte offset
  // D# group 0: count=1, lds_addr, global_addr[56:0], type=2 ("image")
  v4u g0 = { 1u, lo, (unsigned)(ga & 0xffffffffu),
             (unsigned)((ga >> 32) & 0x1ffffffu) | (2u << 30) };
  // D# group 1: data_size=8B; 1-D tile: tensor_dim0 = tile_dim0 = n8 (<=65535)
  v8i g1 = { (int)(3u << 16),            // data_size = 3 (8 bytes)
             (int)((n8 & 0xffffu) << 16),// tensor_dim0[15:0]
             0,                          // tensor_dim0[31:16] | tensor_dim1[15:0]
             (int)((n8 & 0xffffu) << 16),// tile_dim0
             0,                          // tile_dim1 | tile_dim2 (unused)
             (int)n8,                    // tensor_dim0_stride[31:0]
             0, 0 };
  v4i gz4 = { 0, 0, 0, 0 };
  v8i gz8 = { 0, 0, 0, 0, 0, 0, 0, 0 };
  // 6-arg toolchain form: (g0, g1, g2, g3, g4, cpol)
  __builtin_amdgcn_tensor_load_to_lds(g0, g1, gz4, gz4, gz8, 0);
#else
  // fallback: plain wave-0 vector copy (32 lanes x 16B)
  const v8h* s = (const v8h*)gsrc;
  v8h* d = (v8h*)ldst;
  const int n = (int)(n8 >> 1);
  for (int i = (int)(threadIdx.x & 31); i < n; i += 32) d[i] = s[i];
#endif
}

__device__ __forceinline__ void tdm_wait() {
#if __has_builtin(__builtin_amdgcn_tensor_load_to_lds)
  __builtin_amdgcn_s_wait_tensorcnt(0);
#endif
}

// A-fragment (16x32 f16, MxK): lanes 0-15 -> M=lane, K={0..7,16..23};
// lanes 16-31 -> M=lane-16, K={8..15,24..31}.
__device__ __forceinline__ v16h load_a_frag(const _Float16* actw, int kc, int lane) {
  const int m    = lane & 15;
  const int base = kc * 32 + ((lane >> 4) ? 8 : 0);
  const _Float16* p = actw + m * SW + base;
  union { v16h v; v8h h[2]; } u;
  u.h[0] = *(const v8h*)(p);
  u.h[1] = *(const v8h*)(p + 16);
  return u.v;
}

// B-fragment (32x16 f16, KxN): lanes 0-15 -> N=lane, K={0..7,16..23};
// lanes 16-31 -> N=lane-16, K={8..15,24..31}.
__device__ __forceinline__ v16h load_b_frag(const _Float16* wb, int kc, int nt, int lane) {
  const int n  = nt * 16 + (lane & 15);
  const int kb = kc * 32 + ((lane >> 4) ? 8 : 0);
  v16h v;
#pragma unroll
  for (int s = 0; s < 8; ++s) {
    v[s]     = wb[(kb + s) * SW + n];
    v[s + 8] = wb[(kb + 16 + s) * SW + n];
  }
  return v;
}

// C/D (16x16 f32) epilogue: bias + ReLU + f32->f16 back into the wave's act tile.
__device__ __forceinline__ void store_act_tile(_Float16* actw, const v8f& c, int nt,
                                               int lane, const float* bias, float mul) {
  const int n   = lane & 15;
  const int mb  = (lane >> 4) ? 8 : 0;
  const int col = nt * 16 + n;
  const float bb = bias[col];
#pragma unroll
  for (int r = 0; r < 8; ++r) {
    float v = c[r] * mul + bb;
    v = v > 0.0f ? v : 0.0f;
    actw[(mb + r) * SW + col] = (_Float16)v;
  }
}

// ================= main fused kernel: 512 threads = 16 waves ===================
extern "C" __global__ void __launch_bounds__(512)
nerf_hash_mlp(const float* __restrict__ x, const float* __restrict__ table,
              const _Float16* __restrict__ ws,
              const float* __restrict__ b1, const float* __restrict__ b2,
              const float* __restrict__ b3, const float* __restrict__ b4,
              float* __restrict__ out)
{
  __shared__ __align__(16) _Float16 wbuf[256 * SW];        // 135168 B
  __shared__ __align__(16) _Float16 act[16 * 16 * SW];     // 135168 B
  __shared__ float sbias[256];

  const int tid  = threadIdx.x;
  const int lane = tid & 31;
  const int wave = tid >> 5;
  const int row  = lane & 15;
  const int lgrp = lane >> 4;                      // 0: levels 0-7, 1: levels 8-15
  const int pt   = blockIdx.x * 256 + wave * 16 + row;
  _Float16* myact = act + wave * 16 * SW;

  // Issue layer-1 weight DMA first: it streams while we do the hash gathers.
  if (tid < 32) tdm_issue(ws + IMG1_OFF / 2, wbuf, IMG1_BYTES >> 3);

  // ================= Phase 0: hash-grid trilinear encode =================
  {
    const float inv = 1.0f / 1.5f;
    const float px = x[pt * 3 + 0] * inv;
    const float py = x[pt * 3 + 1] * inv;
    const float pz = x[pt * 3 + 2] * inv;
    const float2* tab = (const float2*)table;
#pragma unroll
    for (int li = 0; li < 8; ++li) {
      const int l = lgrp * 8 + li;
      const float sc = c_scale[l];
      const float fx = px * sc + 0.5f, fy = py * sc + 0.5f, fz = pz * sc + 0.5f;
      const float gx = floorf(fx), gy = floorf(fy), gz = floorf(fz);
      const float rx = fx - gx, ry = fy - gy, rz = fz - gz;
      const unsigned ux = (unsigned)gx, uy = (unsigned)gy, uz = (unsigned)gz;
      float f0 = 0.0f, f1 = 0.0f;
#pragma unroll
      for (int c = 0; c < 8; ++c) {
        const unsigned bx = c & 1, by = (c >> 1) & 1, bz = (c >> 2) & 1;
        const unsigned cx = ux + bx, cy = uy + by, cz = uz + bz;
        const float w = (bx ? rx : 1.0f - rx) * (by ? ry : 1.0f - ry)
                      * (bz ? rz : 1.0f - rz);
        unsigned idx;
        if (l < 5) {
          const unsigned r1 = c_res1[l];
          idx = (cx + cy * r1 + cz * r1 * r1) % c_dsize[l];
        } else {
          idx = (cx * 1u) ^ (cy * 2654435761u) ^ (cz * 805459861u);
          idx &= 524287u;
        }
        const float2 tv = tab[idx + c_offs[l]];
        f0 += w * tv.x;
        f1 += w * tv.y;
      }
      myact[row * SW + 2 * l + 0] = (_Float16)(f0 * 256.0f);  // x256 f16 scaling
      myact[row * SW + 2 * l + 1] = (_Float16)(f1 * 256.0f);
    }
  }

  // ================= Layer 1: 32 -> 256 =================
  if (tid < 256) sbias[tid] = b1[tid];
  if (tid < 32) tdm_wait();
  __syncthreads();
  {
    const v16h a0 = load_a_frag(myact, 0, lane);   // preload before overwriting act
#pragma unroll
    for (int nt = 0; nt < 16; ++nt) {
      v8f c = {};
      c = __builtin_amdgcn_wmma_f32_16x16x32_f16(false, a0, false,
            load_b_frag(wbuf, 0, nt, lane), (short)0, c, false, false);
      store_act_tile(myact, c, nt, lane, sbias, 1.0f / 256.0f);
    }
  }

  // ================= Layer 2: 256 -> 256 =================
  __syncthreads();
  if (tid < 256) sbias[tid] = b2[tid];
  if (tid < 32) { tdm_issue(ws + IMG2_OFF / 2, wbuf, IMGW_BYTES >> 3); tdm_wait(); }
  __syncthreads();
  {
    v16h a[8];
#pragma unroll
    for (int kc = 0; kc < 8; ++kc) a[kc] = load_a_frag(myact, kc, lane);
#pragma unroll 2
    for (int nt = 0; nt < 16; ++nt) {
      v8f c = {};
#pragma unroll
      for (int kc = 0; kc < 8; ++kc)
        c = __builtin_amdgcn_wmma_f32_16x16x32_f16(false, a[kc], false,
              load_b_frag(wbuf, kc, nt, lane), (short)0, c, false, false);
      store_act_tile(myact, c, nt, lane, sbias, 1.0f);
    }
  }

  // ================= Layer 3: 256 -> 256 =================
  __syncthreads();
  if (tid < 256) sbias[tid] = b3[tid];
  if (tid < 32) { tdm_issue(ws + IMG3_OFF / 2, wbuf, IMGW_BYTES >> 3); tdm_wait(); }
  __syncthreads();
  {
    v16h a[8];
#pragma unroll
    for (int kc = 0; kc < 8; ++kc) a[kc] = load_a_frag(myact, kc, lane);
#pragma unroll 2
    for (int nt = 0; nt < 16; ++nt) {
      v8f c = {};
#pragma unroll
      for (int kc = 0; kc < 8; ++kc)
        c = __builtin_amdgcn_wmma_f32_16x16x32_f16(false, a[kc], false,
              load_b_frag(wbuf, kc, nt, lane), (short)0, c, false, false);
      store_act_tile(myact, c, nt, lane, sbias, 1.0f);
    }
  }

  // ================= Layer 4: 256 -> 3 (cols padded to 16) + sigmoid =============
  __syncthreads();
  if (tid < 256) sbias[tid] = (tid < 3) ? b4[tid] : 0.0f;
  if (tid < 32) { tdm_issue(ws + IMG4_OFF / 2, wbuf, IMGW_BYTES >> 3); tdm_wait(); }
  __syncthreads();
  {
    v16h a[8];
#pragma unroll
    for (int kc = 0; kc < 8; ++kc) a[kc] = load_a_frag(myact, kc, lane);
    v8f c = {};
#pragma unroll
    for (int kc = 0; kc < 8; ++kc)
      c = __builtin_amdgcn_wmma_f32_16x16x32_f16(false, a[kc], false,
            load_b_frag(wbuf, kc, 0, lane), (short)0, c, false, false);
    const int n = lane & 15;
    if (n < 3) {
      const int mb = lgrp ? 8 : 0;
      const float bb = sbias[n];
#pragma unroll
      for (int r = 0; r < 8; ++r) {
        float v = c[r] + bb;
        v = 1.0f / (1.0f + __expf(-v));
        out[(blockIdx.x * 256 + wave * 16 + mb + r) * 3 + n] = v;
      }
    }
  }
}

extern "C" void kernel_launch(void* const* d_in, const int* in_sizes, int n_in,
                              void* d_out, int out_size, void* d_ws, size_t ws_size,
                              hipStream_t stream) {
  const float* x  = (const float*)d_in[0];
  const float* tb = (const float*)d_in[1];
  const float* w1 = (const float*)d_in[2];
  const float* b1 = (const float*)d_in[3];
  const float* w2 = (const float*)d_in[4];
  const float* b2 = (const float*)d_in[5];
  const float* w3 = (const float*)d_in[6];
  const float* b3 = (const float*)d_in[7];
  const float* w4 = (const float*)d_in[8];
  const float* b4 = (const float*)d_in[9];
  float* out = (float*)d_out;
  _Float16* ws = (_Float16*)d_ws;

  // 1) build padded f16 weight images in d_ws (143360 elements)
  hipLaunchKernelGGL(prep_weights, dim3(560), dim3(256), 0, stream,
                     w1, w2, w3, w4, ws);
  // 2) fused hash-encode + MLP: 256 points per 512-thread block
  hipLaunchKernelGGL(nerf_hash_mlp, dim3(524288 / 256), dim3(512), 0, stream,
                     x, tb, (const _Float16*)ws, b1, b2, b3, b4, out);

  (void)in_sizes; (void)n_in; (void)out_size; (void)ws_size;
}
